// Edge_sampler_46033459479299
// MI455X (gfx1250) — matrix operand order repaired and checked by reading, not compiled, not run on hardware
//
#include <hip/hip_runtime.h>
#include <stdint.h>

// ---- problem constants (match reference file) ----
#define NN   50000
#define EE   800000
#define TT   (2*EE)       // 1,600,000 total edges (divisible by 256)
#define NF   128          // 2*HID
#define NB   50
#define LN_EPS 1e-5f

typedef __attribute__((ext_vector_type(16))) _Float16 v16h;
typedef __attribute__((ext_vector_type(8)))  float    v8f;

// ---- workspace layout (bytes) ----
#define WS_COUNTS_ALL 0                     // 64 ints
#define WS_COUNTS_POS 256                   // 64 ints
#define WS_STARTS     512                   // 64 ints
#define WS_K          768                   // 64 ints
#define WS_PACKW1     1024                  // 8 tiles * 2 ksteps * 512 halfs = 16 KB
#define WS_ORDER      32768                 // TT ints
#define WS_LOGITS     (32768 + TT*4)        // TT floats

// ------------------------------------------------------------------
// shfl-based block inclusive scan (256 threads = 8 waves): 3 barriers total.
// After return, warpTotals[7] holds the block total (valid until next call).
__device__ inline int block_scan_incl(int val, int* warpTotals, int tid) {
  int lane = tid & 31, w = tid >> 5;
  __syncthreads();                           // protect warpTotals reuse
#pragma unroll
  for (int off = 1; off < 32; off <<= 1) {
    int n = __shfl_up(val, off, 32);
    if (lane >= off) val += n;
  }
  if (lane == 31) warpTotals[w] = val;
  __syncthreads();
  if (w == 0) {
    int t = (lane < 8) ? warpTotals[lane] : 0;
#pragma unroll
    for (int off = 1; off < 8; off <<= 1) {
      int n = __shfl_up(t, off, 32);
      if (lane >= off) t += n;
    }
    if (lane < 8) warpTotals[lane] = t;      // inclusive wave totals
  }
  __syncthreads();
  return val + ((w > 0) ? warpTotals[w - 1] : 0);
}

// ------------------------------------------------------------------
// 0) zero node_mask output region + segment counters
__global__ void zero_init(int* counters, float* node_mask) {
  int i = blockIdx.x * blockDim.x + threadIdx.x;
  if (i < 128) counters[i] = 0;            // countsAll[64] + countsPos[64] contiguous
  if (i < NN)  node_mask[i] = 0.0f;
}

// ------------------------------------------------------------------
// 1) pack W1 (rows 0..63; the conn row 64 is folded into the epilogue)
//    into per-lane WMMA B-operand layout: lane l holds column N = nt*16+(l&15),
//    element e holds K = s*32 + (e<8 ? (l>>4)*8 + e : 16 + (l>>4)*8 + (e-8)).
__global__ void pack_w1(const float* __restrict__ W1, _Float16* __restrict__ packed) {
  int nt = blockIdx.x, s = blockIdx.y, l = threadIdx.x;
  int nf = nt * 16 + (l & 15);
  int hb = l >> 4;
  _Float16* dst = packed + (nt * 2 + s) * 512 + l * 16;
#pragma unroll
  for (int e = 0; e < 16; ++e) {
    int k = s * 32 + (e < 8 ? hb * 8 + e : 16 + hb * 8 + (e - 8));          // k in [0,64)
    dst[e] = (_Float16)W1[k * NF + nf];
  }
}

// ------------------------------------------------------------------
// 2) per-segment edge histograms
__global__ void count_seg(const int* __restrict__ pos_e, const int* __restrict__ neg_e,
                          const int* __restrict__ node_batch,
                          int* countsAll, int* countsPos) {
  int j = blockIdx.x * blockDim.x + threadIdx.x;
  if (j >= TT) return;
  int src = (j < EE) ? pos_e[j] : neg_e[j - EE];
  int g = node_batch[src];
  atomicAdd(&countsAll[g], 1);
  if (j < EE) atomicAdd(&countsPos[g], 1);
}

// 3) tiny scan: starts = exclusive cumsum(countsAll); k = floor(0.9*countsPos)
__global__ void scan_seg(const int* countsAll, const int* countsPos, int* starts, int* kk) {
  if (threadIdx.x == 0 && blockIdx.x == 0) {
    int cum = 0;
    for (int g = 0; g < NB; ++g) {
      starts[g] = cum; cum += countsAll[g];
      kk[g] = (int)floorf((float)countsPos[g] * 0.9f);
    }
  }
}

// ------------------------------------------------------------------
// 4) stable counting-sort rank: one block per segment, ordered block scan
//    reproduces jnp.argsort(stable) by segment: order[pos] = original edge idx.
#define RB 256
__global__ void stable_rank(const int* __restrict__ pos_e, const int* __restrict__ neg_e,
                            const int* __restrict__ node_batch,
                            const int* __restrict__ starts, int* __restrict__ order) {
  __shared__ int wtot[8];
  __shared__ int running;
  int g = blockIdx.x, tid = threadIdx.x;
  if (tid == 0) running = starts[g];
  for (int base = 0; base < TT; base += RB) {           // TT % RB == 0
    int j = base + tid;
    int src = (j < EE) ? pos_e[j] : neg_e[j - EE];
    int flag = (node_batch[src] == g) ? 1 : 0;
    int incl = block_scan_incl(flag, wtot, tid);        // includes leading barrier
    int cur = running;
    if (flag) order[cur + (incl - flag)] = j;
    __syncthreads();
    if (tid == 0) running += wtot[7];
  }
}

// ------------------------------------------------------------------
// 5) fused MLP via WMMA. One wave handles 16 edges; A = enc (16x64 f16),
//    B = packed W1 tiles; b1 + conn column folded into the epilogue
//    (before LayerNorm stats) so the WMMA loop carries no extra live values.
template <int BASE>
__device__ inline void put8(v16h& a, float4 s0, float4 s1, float4 d0, float4 d1) {
  a[BASE + 0] = (_Float16)(s0.x + d0.x); a[BASE + 1] = (_Float16)(s0.y + d0.y);
  a[BASE + 2] = (_Float16)(s0.z + d0.z); a[BASE + 3] = (_Float16)(s0.w + d0.w);
  a[BASE + 4] = (_Float16)(s1.x + d1.x); a[BASE + 5] = (_Float16)(s1.y + d1.y);
  a[BASE + 6] = (_Float16)(s1.z + d1.z); a[BASE + 7] = (_Float16)(s1.w + d1.w);
}

__global__ void fused_mlp_logits(const float* __restrict__ h,
                                 const float* __restrict__ W1,
                                 const float* __restrict__ b1,
                                 const float* __restrict__ ln_g,
                                 const float* __restrict__ ln_b,
                                 const float* __restrict__ W2,
                                 const float* __restrict__ b2,
                                 const int* __restrict__ pos_e,
                                 const int* __restrict__ neg_e,
                                 const int* __restrict__ order,
                                 const _Float16* __restrict__ packW1,
                                 float* __restrict__ logits) {
  const int lane = threadIdx.x & 31;
  const int wave = threadIdx.x >> 5;
  const int half = lane >> 4;       // A: lanes 0-15 hold K 0..7/16..23, 16-31 hold 8..15/24..31
  const int r    = lane & 15;       // A row (edge) for this lane
  const int tileBase = (blockIdx.x * (blockDim.x >> 5) + wave) * 16;

  int j = order[tileBase + r];
  int src, dst;
  if (j < EE) { src = pos_e[j];      dst = pos_e[j + EE]; }
  else        { src = neg_e[j - EE]; dst = neg_e[j]; }
  float conn = (j < EE) ? 1.0f : 0.0f;

  const float4* hs = (const float4*)h + (size_t)src * 16;
  const float4* hd = (const float4*)h + (size_t)dst * 16;

  v16h a0, a1;
  // k-step 0: enc[half*8 .. +8) and enc[16+half*8 .. +8)
  put8<0>(a0, hs[half*2 + 0],  hs[half*2 + 1],  hd[half*2 + 0],  hd[half*2 + 1]);
  put8<8>(a0, hs[4 + half*2],  hs[5 + half*2],  hd[4 + half*2],  hd[5 + half*2]);
  // k-step 1: +32 floats
  put8<0>(a1, hs[8 + half*2],  hs[9 + half*2],  hd[8 + half*2],  hd[9 + half*2]);
  put8<8>(a1, hs[12 + half*2], hs[13 + half*2], hd[12 + half*2], hd[13 + half*2]);

  const int nfBase = lane & 15;     // C/D layout: lane = N within tile
  v8f c[8];
#pragma unroll
  for (int nt = 0; nt < 8; ++nt)
#pragma unroll
    for (int v = 0; v < 8; ++v) c[nt][v] = 0.0f;

#pragma unroll
  for (int nt = 0; nt < 8; ++nt) {
    v16h bm0 = *(const v16h*)(packW1 + (nt * 2 + 0) * 512 + lane * 16);
    v16h bm1 = *(const v16h*)(packW1 + (nt * 2 + 1) * 512 + lane * 16);
    c[nt] = __builtin_amdgcn_wmma_f32_16x16x32_f16(false, a0, false, bm0,
                                                   (short)0, c[nt], false, false);
    c[nt] = __builtin_amdgcn_wmma_f32_16x16x32_f16(false, a1, false, bm1,
                                                   (short)0, c[nt], false, false);
  }

  // ---- epilogue: fold bias + conn column, then LayerNorm + ReLU + W2 ----
  // broadcast conn per output row M = half*8 + v (row M lives in lane M%16's `conn`)
  float connM[8];
#pragma unroll
  for (int v = 0; v < 8; ++v) connM[v] = __shfl(conn, half * 8 + v, 32);

#pragma unroll
  for (int nt = 0; nt < 8; ++nt) {
    int f = nt * 16 + nfBase;
    float bb  = b1[f];
    float w64 = W1[64 * NF + f];    // conn row of W1 (rank-1 update)
#pragma unroll
    for (int v = 0; v < 8; ++v) c[nt][v] += bb + connM[v] * w64;
  }

  // LayerNorm stats per row M = half*8 + v: reduce across the 16 lanes of this half
  float s1[8], s2[8];
#pragma unroll
  for (int v = 0; v < 8; ++v) { s1[v] = 0.f; s2[v] = 0.f; }
#pragma unroll
  for (int nt = 0; nt < 8; ++nt)
#pragma unroll
    for (int v = 0; v < 8; ++v) { float x = c[nt][v]; s1[v] += x; s2[v] += x * x; }
#pragma unroll
  for (int m = 1; m < 16; m <<= 1)
#pragma unroll
    for (int v = 0; v < 8; ++v) {
      s1[v] += __shfl_xor(s1[v], m, 32);
      s2[v] += __shfl_xor(s2[v], m, 32);
    }
  float mu[8], rstd[8];
#pragma unroll
  for (int v = 0; v < 8; ++v) {
    mu[v] = s1[v] * (1.0f / NF);
    float var = s2[v] * (1.0f / NF) - mu[v] * mu[v];
    rstd[v] = rsqrtf(var + LN_EPS);
  }

  // normalize + ReLU + dot with W2
  float lp[8];
#pragma unroll
  for (int v = 0; v < 8; ++v) lp[v] = 0.f;
#pragma unroll
  for (int nt = 0; nt < 8; ++nt) {
    int f = nt * 16 + nfBase;
    float gg = ln_g[f], bbb = ln_b[f], ww = W2[f];
#pragma unroll
    for (int v = 0; v < 8; ++v) {
      float xn = (c[nt][v] - mu[v]) * rstd[v] * gg + bbb;
      lp[v] += fmaxf(xn, 0.f) * ww;
    }
  }
#pragma unroll
  for (int m = 1; m < 16; m <<= 1)
#pragma unroll
    for (int v = 0; v < 8; ++v) lp[v] += __shfl_xor(lp[v], m, 32);

  if (nfBase < 8) {                 // lane e of each half writes row M = half*8 + e
    float out = lp[0];
#pragma unroll
    for (int v = 1; v < 8; ++v) out = (nfBase == v) ? lp[v] : out;
    logits[tileBase + half * 8 + nfBase] = out + b2[0];
  }
}

// ------------------------------------------------------------------
// 6) per-segment top-k: 4-pass 8-bit radix select on monotone keys, then an
//    ordered pass with stable tie-breaking (matches lexsort stability).
#define TKB 256
__device__ inline unsigned fkey(float f) {
  unsigned u = __float_as_uint(f);
  return (u & 0x80000000u) ? ~u : (u | 0x80000000u);   // larger key = larger float
}

__global__ void topk_select(const float* __restrict__ logits,
                            const int* __restrict__ countsAll,
                            const int* __restrict__ starts,
                            const int* __restrict__ kk,
                            const int* __restrict__ order,
                            const int* __restrict__ pos_e, const int* __restrict__ neg_e,
                            float* __restrict__ mask_out, float* __restrict__ ew_out,
                            float* __restrict__ node_mask) {
  __shared__ unsigned hist[256];
  __shared__ unsigned sPrefix;
  __shared__ int sRem;
  __shared__ int wtot[8];
  __shared__ int sTieBase;

  int g = blockIdx.x, tid = threadIdx.x;
  int n = countsAll[g], base = starts[g], k = kk[g];
  bool selAll  = (k >= n);
  bool selNone = (k <= 0);
  unsigned thresh = 0; int tiesTake = 0;

  if (!selAll && !selNone) {
    if (tid == 0) { sPrefix = 0; sRem = k; }
    __syncthreads();
    for (int pass = 0; pass < 4; ++pass) {
      int shift = 24 - 8 * pass;
      hist[tid] = 0;
      __syncthreads();
      unsigned pref = sPrefix; int rem = sRem;
      for (int i = tid; i < n; i += TKB) {
        unsigned u = fkey(logits[base + i]);
        bool match = (pass == 0) || ((u >> (shift + 8)) == pref);
        if (match) atomicAdd(&hist[(u >> shift) & 255u], 1u);
      }
      __syncthreads();
      if (tid == 0) {
        int cum = 0, b = 255;
        for (; b > 0; --b) {                 // descending: find k-th bin
          int cb = (int)hist[b];
          if (cum + cb >= rem) break;
          cum += cb;
        }
        sPrefix = (pref << 8) | (unsigned)b;
        sRem = rem - cum;
      }
      __syncthreads();
    }
    thresh = sPrefix; tiesTake = sRem;
  }

  if (tid == 0) sTieBase = 0;
  for (int cb = 0; cb < n; cb += TKB) {
    int i = cb + tid;
    int sel = 0, eq = 0; float lg = 0.f;
    if (i < n) {
      lg = logits[base + i];
      if (selAll) sel = 1;
      else if (!selNone) {
        unsigned u = fkey(lg);
        if (u > thresh) sel = 1;
        else if (u == thresh) eq = 1;
      }
    }
    int incl = block_scan_incl(eq, wtot, tid);           // includes leading barrier
    int tb = sTieBase;
    if (eq && (tb + (incl - eq)) < tiesTake) sel = 1;    // first ties in index order win
    if (i < n) {
      int p = base + i;
      mask_out[p] = sel ? 1.0f : 0.0f;
      ew_out[p]   = sel ? lg : 0.0f;
      if (sel) {
        int jj = order[p];
        int s, d;
        if (jj < EE) { s = pos_e[jj];      d = pos_e[jj + EE]; }
        else         { s = neg_e[jj - EE]; d = neg_e[jj]; }
        node_mask[s] = 1.0f;             // idempotent stores; races benign
        node_mask[d] = 1.0f;
      }
    }
    __syncthreads();
    if (tid == 0) sTieBase += wtot[7];
  }
}

// ------------------------------------------------------------------
extern "C" void kernel_launch(void* const* d_in, const int* in_sizes, int n_in,
                              void* d_out, int out_size, void* d_ws, size_t ws_size,
                              hipStream_t stream) {
  (void)in_sizes; (void)n_in; (void)out_size; (void)ws_size;
  const float* h    = (const float*)d_in[0];
  const float* W1   = (const float*)d_in[1];
  const float* b1   = (const float*)d_in[2];
  const float* ln_g = (const float*)d_in[3];
  const float* ln_b = (const float*)d_in[4];
  const float* W2   = (const float*)d_in[5];
  const float* b2   = (const float*)d_in[6];
  const int* pos_e  = (const int*)d_in[7];       // (2,E) flat: [0,E)=src, [E,2E)=dst
  const int* neg_e  = (const int*)d_in[8];
  const int* node_batch = (const int*)d_in[9];
  // d_in[10] = num_graphs == NB (compile-time constant, matches reference)

  char* ws = (char*)d_ws;
  int*      countsAll = (int*)(ws + WS_COUNTS_ALL);
  int*      countsPos = (int*)(ws + WS_COUNTS_POS);
  int*      starts    = (int*)(ws + WS_STARTS);
  int*      kk        = (int*)(ws + WS_K);
  _Float16* packW1    = (_Float16*)(ws + WS_PACKW1);
  int*      order     = (int*)(ws + WS_ORDER);
  float*    logits    = (float*)(ws + WS_LOGITS);

  float* mask_out  = (float*)d_out;      // T bools-as-floats
  float* ew_out    = mask_out + TT;      // T floats
  float* node_mask = ew_out + TT;        // N bools-as-floats

  zero_init<<<(NN + 255) / 256, 256, 0, stream>>>(countsAll, node_mask);
  pack_w1<<<dim3(8, 2), 32, 0, stream>>>(W1, packW1);
  count_seg<<<(TT + 255) / 256, 256, 0, stream>>>(pos_e, neg_e, node_batch,
                                                  countsAll, countsPos);
  scan_seg<<<1, 32, 0, stream>>>(countsAll, countsPos, starts, kk);
  stable_rank<<<NB, RB, 0, stream>>>(pos_e, neg_e, node_batch, starts, order);
  fused_mlp_logits<<<TT / 128, 256, 0, stream>>>(h, W1, b1, ln_g, ln_b, W2, b2,
                                                 pos_e, neg_e, order, packW1, logits);
  topk_select<<<NB, TKB, 0, stream>>>(logits, countsAll, starts, kk, order,
                                      pos_e, neg_e, mask_out, ew_out, node_mask);
}
